// SequentialMNIST_35510789603609
// MI455X (gfx1250) — matrix operand order repaired and checked by reading, not compiled, not run on hardware
//
#include <hip/hip_runtime.h>

// ---------------------------------------------------------------------------
// S4D sequential-MNIST forward for MI455X (gfx1250, wave32, WMMA).
// FFT conv re-expressed as per-channel lower-triangular block-Toeplitz GEMM
// executed with v_wmma_f32_16x16x32_bf16 (bf16 inputs, fp32 accumulate).
// K staging uses global_load_async_to_lds_b32 (ASYNCcnt) per CDNA5 ISA ch.8.
// ---------------------------------------------------------------------------

#define HDIM   512
#define NSTATE 64
#define NLAYER 6
#define LSEQ   784
#define BDIM   64
#define LB     (LSEQ * BDIM)          // 50176
#define HLB    ((size_t)HDIM * LB)    // 25,690,112 elements
#define NTILES 49                     // LSEQ/16 row tiles
#define UTSTRIDE 72                   // sUT row stride in halfs (144B, 16B-mult)

typedef __attribute__((ext_vector_type(16))) __bf16 v16bf;
typedef __attribute__((ext_vector_type(8)))  float  v8f;
typedef __attribute__((ext_vector_type(4)))  unsigned int v4u;

__device__ __forceinline__ unsigned short f2bf(float f) {
    unsigned int u = __float_as_uint(f);
    unsigned int r = u + 0x7FFFu + ((u >> 16) & 1u);   // round-to-nearest-even
    return (unsigned short)(r >> 16);
}

// ------------------------- encoder: h = x*enc_w + enc_b ---------------------
__global__ void s4d_encode(const float* __restrict__ x,     // (B,L,1)
                           const float* __restrict__ ew,    // (1,H)
                           const float* __restrict__ eb,    // (H,)
                           float* __restrict__ hf,          // [H][L][B]
                           unsigned short* __restrict__ hb) // bf16 copy
{
    size_t t = (size_t)blockIdx.x * blockDim.x + threadIdx.x;
    if (t >= HLB) return;
    int b = (int)(t & 63);
    size_t r = t >> 6;
    int l  = (int)(r % LSEQ);
    int hh = (int)(r / LSEQ);
    float v = x[(size_t)b * LSEQ + l] * ew[hh] + eb[hh];
    hf[t] = v;
    hb[t] = f2bf(v);
}

// --------- per-(h,n) discretization:  w = dt*A,  Cd = C*(exp(w)-1)/A --------
__global__ void s4d_cdparams(const float* __restrict__ log_dt,     // (H,) slice
                             const float* __restrict__ log_A_real, // (H,N) slice
                             const float* __restrict__ A_imag,
                             const float* __restrict__ C_re,
                             const float* __restrict__ C_im,
                             float* __restrict__ wre, float* __restrict__ wim,
                             float* __restrict__ cre, float* __restrict__ cim)
{
    int t = blockIdx.x * blockDim.x + threadIdx.x;     // h*N + n
    if (t >= HDIM * NSTATE) return;
    int hh = t >> 6;
    float dt  = __expf(log_dt[hh]);
    float Ar  = -__expf(log_A_real[t]);
    float Ai  = A_imag[t];
    float dre = Ar * dt, dimg = Ai * dt;
    float e   = __expf(dre);
    float sn, cs; __sincosf(dimg, &sn, &cs);
    float Ere = e * cs - 1.f, Eim = e * sn;            // exp(dtA) - 1
    float Cr = C_re[t], Ci = C_im[t];
    float CEr = Cr * Ere - Ci * Eim;
    float CEi = Cr * Eim + Ci * Ere;
    float inv = 1.f / (Ar * Ar + Ai * Ai);             // divide by A (complex)
    wre[t] = dre;  wim[t] = dimg;
    cre[t] = (CEr * Ar + CEi * Ai) * inv;
    cim[t] = (CEi * Ar - CEr * Ai) * inv;
}

// --------------- K[h,l] = 2*Re( sum_n Cd[h,n] * exp(w[h,n]*l) ) -------------
__global__ void s4d_kgen(const float* __restrict__ wre, const float* __restrict__ wim,
                         const float* __restrict__ cre, const float* __restrict__ cim,
                         unsigned short* __restrict__ Kbf)          // [H][L] bf16
{
    int t = blockIdx.x * blockDim.x + threadIdx.x;     // h*L + l
    if (t >= HDIM * LSEQ) return;
    int l  = t % LSEQ;
    int hh = t / LSEQ;
    float lf = (float)l;
    float acc = 0.f;
    int base = hh * NSTATE;
    #pragma unroll 4
    for (int n = 0; n < NSTATE; ++n) {
        int idx = base + n;                            // broadcast across lanes
        float e = __expf(wre[idx] * lf);
        float sn, cs; __sincosf(wim[idx] * lf, &sn, &cs);
        acc += cre[idx] * (e * cs) - cim[idx] * (e * sn);
    }
    Kbf[t] = f2bf(2.f * acc);
}

// ------- causal depthwise conv as block-Toeplitz WMMA: z = K*u + D*u --------
// grid = (13 tile-quads, 512 channels), 128 threads = 4 wave32.
// Wave w owns row tile (4*blockIdx.x + w) and computes ALL 4 batch tiles with
// one shared A fragment (Toeplitz of K). B fragments come from a transposed
// LDS slab as two aligned b128 loads per tile.
__global__ __launch_bounds__(128) void s4d_conv_wmma(
    const unsigned short* __restrict__ hb,   // bf16 [H][L][B]
    const unsigned short* __restrict__ Kbf,  // bf16 [H][L]
    const float* __restrict__ hf,            // fp32 [H][L][B] (for D-term)
    const float* __restrict__ Dv,            // (H,) slice
    float* __restrict__ z)                   // fp32 [H][L][B]
{
    __shared__ __align__(16) unsigned short sK[LSEQ];             // K rows
    __shared__ __align__(16) unsigned short sUT[BDIM * UTSTRIDE]; // [b][k] slab

    const int i0   = blockIdx.x * 4;
    const int h    = blockIdx.y;
    const int t    = threadIdx.x;
    const int wave = t >> 5;
    const int lane = t & 31;
    const int grp  = lane >> 4;              // lane group 0/1
    const int mn   = lane & 15;              // M (for A) / N (for B)

    const int  iw    = i0 + wave;            // this wave's row tile
    const bool live  = (iw < NTILES);
    const int  iTop  = (i0 + 3 < NTILES - 1) ? (i0 + 3) : (NTILES - 1);
    const int  blkHi = iTop * 16 + 15;       // highest row any wave needs
    const int  jmaxW = live ? ((iw * 16 + 15) >> 5) : -1;
    const int  jmaxB = blkHi >> 5;

    // ---- async stage K[h][0..blkHi] into sK (global_load_async_to_lds_b32) --
    {
        const unsigned int* gsrc = (const unsigned int*)(Kbf + (size_t)h * LSEQ);
        unsigned int ldsK = (unsigned int)(size_t)&sK[0];
        const int n32 = (blkHi + 2) >> 1;    // dwords to copy
        for (int k = t; k < n32; k += 128) {
            unsigned int boff = (unsigned int)k * 4u;
            asm volatile("global_load_async_to_lds_b32 %0, %1, %2"
                         :
                         : "v"(ldsK + boff), "v"(boff), "s"(gsrc)
                         : "memory");
        }
        asm volatile("s_wait_asynccnt 0x0" ::: "memory");
    }
    __syncthreads();

    v8f acc[4];
    #pragma unroll
    for (int bt = 0; bt < 4; ++bt)
        acc[bt] = (v8f){0.f,0.f,0.f,0.f,0.f,0.f,0.f,0.f};

    const size_t ubase = (size_t)h * LB;

    for (int j = 0; j <= jmaxB; ++j) {
        if (j < jmaxB)                       // global_prefetch_b8 next slab
            __builtin_prefetch(hb + ubase + (size_t)((j + 1) * 32) * BDIM, 0, 0);
        __syncthreads();                     // sUT reuse fence
        { // transpose-stage rows [j*32, j*32+32) x 64 cols: sUT[b][k]
            const unsigned int* src =
                (const unsigned int*)(hb + ubase + (size_t)(j * 32) * BDIM);
            const int maxrow = LSEQ - j * 32;
            for (int k32 = t; k32 < 1024; k32 += 128) {
                int row = k32 >> 5;          // 0..31
                int bp  = (k32 & 31) << 1;   // batch pair
                unsigned int vv = (row < maxrow) ? src[k32] : 0u;
                sUT[(size_t)bp       * UTSTRIDE + row] = (unsigned short)(vv & 0xFFFFu);
                sUT[(size_t)(bp + 1) * UTSTRIDE + row] = (unsigned short)(vv >> 16);
            }
        }
        __syncthreads();

        if (!live || j > jmaxW) continue;    // uniform barrier count per j

        // A fragment (16x32 bf16 Toeplitz of K), built once per wave per chunk
        union { v16bf v; unsigned short u[16]; } Af;
        const int base_l = iw * 16 + mn;
        #pragma unroll
        for (int e = 0; e < 16; ++e) {
            int v = e >> 1, p = e & 1;
            int klA = v * 2 + p + grp * 16;
            int d = base_l - (j * 32 + klA); // A[m,k] = K[l - k], 0 above diag
            Af.u[e] = (d >= 0) ? sK[d] : (unsigned short)0;
        }

        // 4 batch tiles reuse Af; B = 32 contiguous bytes -> 2x ds_load_b128
        #pragma unroll
        for (int bt = 0; bt < 4; ++bt) {
            union { v16bf v; v4u q[2]; } Bf;
            const v4u* qp = (const v4u*)(sUT + (size_t)(bt * 16 + mn) * UTSTRIDE
                                             + grp * 16);
            Bf.q[0] = qp[0];
            Bf.q[1] = qp[1];
            acc[bt] = __builtin_amdgcn_wmma_f32_16x16x32_bf16(
                          false, Af.v, false, Bf.v, (short)0, acc[bt],
                          false, false);
        }
    }

    if (!live) return;
    // C/D layout: VGPR r -> M = r + 8*grp, lane%16 -> N. Fuse skip term u*D.
    const float Dh = Dv[h];
    #pragma unroll
    for (int bt = 0; bt < 4; ++bt) {
        #pragma unroll
        for (int r = 0; r < 8; ++r) {
            int l = iw * 16 + r + grp * 8;
            size_t off = ubase + (size_t)l * BDIM + bt * 16 + mn;
            z[off] = acc[bt][r] + Dh * hf[off];
        }
    }
}

// ----------------- residual + LayerNorm over H (eps = 1e-5) -----------------
__global__ void s4d_resln(float* __restrict__ hf, const float* __restrict__ z,
                          unsigned short* __restrict__ hb,
                          const float* __restrict__ g, const float* __restrict__ bia,
                          int use_res)
{
    int t = blockIdx.x * blockDim.x + threadIdx.x;     // l*64 + b
    if (t >= LB) return;
    float s = 0.f, s2 = 0.f;
    for (int hh = 0; hh < HDIM; ++hh) {
        size_t off = (size_t)hh * LB + t;
        float v = hf[off] + (use_res ? z[off] : 0.f);
        s += v; s2 += v * v;
    }
    const float mu  = s * (1.f / HDIM);
    const float var = s2 * (1.f / HDIM) - mu * mu;
    const float inv = rsqrtf(var + 1e-5f);
    for (int hh = 0; hh < HDIM; ++hh) {
        size_t off = (size_t)hh * LB + t;
        float v = hf[off] + (use_res ? z[off] : 0.f);
        float o = (v - mu) * inv * g[hh] + bia[hh];
        hf[off] = o;
        hb[off] = f2bf(o);
    }
}

// --------------------------- mean pool over L -------------------------------
__global__ void s4d_pool(const float* __restrict__ hf, float* __restrict__ pooled)
{
    int t = blockIdx.x * blockDim.x + threadIdx.x;     // h*64 + b
    if (t >= HDIM * BDIM) return;
    int b = t & 63, hh = t >> 6;
    float s = 0.f;
    size_t base = (size_t)hh * LB + b;
    for (int l = 0; l < LSEQ; ++l) s += hf[base + (size_t)l * BDIM];
    pooled[t] = s * (1.f / LSEQ);
}

// ------------------------------- decoder ------------------------------------
__global__ void s4d_decode(const float* __restrict__ pooled,
                           const float* __restrict__ dw,  // (H,10)
                           const float* __restrict__ db,  // (10,)
                           float* __restrict__ out)       // (B,10)
{
    int t = blockIdx.x * blockDim.x + threadIdx.x;
    if (t >= BDIM * 10) return;
    int b = t / 10, o = t % 10;
    float s = db[o];
    for (int hh = 0; hh < HDIM; ++hh)
        s += pooled[hh * BDIM + b] * dw[hh * 10 + o];
    out[b * 10 + o] = s;
}

// ---------------------------------------------------------------------------
extern "C" void kernel_launch(void* const* d_in, const int* in_sizes, int n_in,
                              void* d_out, int out_size, void* d_ws, size_t ws_size,
                              hipStream_t stream)
{
    const float* x     = (const float*)d_in[0];
    const float* enc_w = (const float*)d_in[1];
    const float* enc_b = (const float*)d_in[2];
    const float* log_dt     = (const float*)d_in[3];   // (NL,H)
    const float* log_A_real = (const float*)d_in[4];   // (NL,H,N)
    const float* A_imag     = (const float*)d_in[5];
    const float* C_re       = (const float*)d_in[6];
    const float* C_im       = (const float*)d_in[7];
    const float* Dp         = (const float*)d_in[8];   // (NL,H)
    const float* ln_g       = (const float*)d_in[9];
    const float* ln_b       = (const float*)d_in[10];
    const float* fn_g       = (const float*)d_in[11];
    const float* fn_b       = (const float*)d_in[12];
    const float* dec_w      = (const float*)d_in[13];
    const float* dec_b      = (const float*)d_in[14];
    float* out = (float*)d_out;

    // workspace partition (256B aligned slices)
    char* p = (char*)d_ws;
    size_t used = 0;
    auto alloc = [&](size_t bytes) {
        char* r = p + used;
        used += (bytes + 255) & ~(size_t)255;
        return (void*)r;
    };
    float*          hf  = (float*)         alloc(HLB * sizeof(float));
    float*          zb  = (float*)         alloc(HLB * sizeof(float));
    unsigned short* hbb = (unsigned short*)alloc(HLB * sizeof(unsigned short));
    unsigned short* Kbf = (unsigned short*)alloc((size_t)HDIM * LSEQ * 2);
    float* wre = (float*)alloc(HDIM * NSTATE * 4);
    float* wim = (float*)alloc(HDIM * NSTATE * 4);
    float* cre = (float*)alloc(HDIM * NSTATE * 4);
    float* cim = (float*)alloc(HDIM * NSTATE * 4);
    float* pooled = (float*)alloc(HDIM * BDIM * 4);
    if (used > ws_size) return;   // workspace too small: bail deterministically

    // encoder
    {
        int blk = 256;
        unsigned nb = (unsigned)((HLB + blk - 1) / blk);
        s4d_encode<<<dim3(nb), blk, 0, stream>>>(x, enc_w, enc_b, hf, hbb);
    }

    for (int layer = 0; layer < NLAYER; ++layer) {
        const size_t oH  = (size_t)layer * HDIM;
        const size_t oHN = (size_t)layer * HDIM * NSTATE;

        s4d_cdparams<<<(HDIM * NSTATE + 255) / 256, 256, 0, stream>>>(
            log_dt + oH, log_A_real + oHN, A_imag + oHN,
            C_re + oHN, C_im + oHN, wre, wim, cre, cim);

        s4d_kgen<<<(HDIM * LSEQ + 255) / 256, 256, 0, stream>>>(
            wre, wim, cre, cim, Kbf);

        s4d_conv_wmma<<<dim3((NTILES + 3) / 4, HDIM), 128, 0, stream>>>(
            hbb, Kbf, hf, Dp + oH, zb);

        s4d_resln<<<(LB + 255) / 256, 256, 0, stream>>>(
            hf, zb, hbb, ln_g + oH, ln_b + oH, 1);
    }

    // final LN (no residual), pool, decode
    s4d_resln<<<(LB + 255) / 256, 256, 0, stream>>>(hf, zb, hbb, fn_g, fn_b, 0);
    s4d_pool<<<(HDIM * BDIM + 255) / 256, 256, 0, stream>>>(hf, pooled);
    s4d_decode<<<(BDIM * 10 + 255) / 256, 256, 0, stream>>>(pooled, dec_w, dec_b, out);
}
